// MiniQuixerPQCBlock_15109694947364
// MI455X (gfx1250) — matrix operand order repaired and checked by reading, not compiled, not run on hardware
//
#include <hip/hip_runtime.h>
#include <hip/hip_fp16.h>

typedef __attribute__((ext_vector_type(16))) _Float16 v16h;
typedef __attribute__((ext_vector_type(8)))  float    v8f;

#define B_TOT    512
#define DCTX     256
#define NQ       16
#define NLAYER   4
#define NSTATE   65536      // 2^16 amplitudes
#define NPAIR    32768
#define NTHREADS 1024
#define PI_F     3.14159265358979323846f

// ---------------------------------------------------------------------------
// Kernel 1: angles = pi * tanh(context @ W^T + b)   [512,16]  via WMMA f16
// One wave (32 threads) computes a 16x16 output tile; K-loop of 8 x (K=32).
// ---------------------------------------------------------------------------
__global__ __launch_bounds__(32)
void angles_wmma_kernel(const float* __restrict__ ctx,   // [512,256]
                        const float* __restrict__ W,     // [16,256]
                        const float* __restrict__ bias,  // [16]
                        float* __restrict__ angles)      // [512,16]
{
  const int lane = threadIdx.x;
  const int m0   = blockIdx.x * 16;          // 32 blocks * 16 rows = 512
  const int Mrow = lane & 15;
  const int n    = lane & 15;
  const int aKb  = (lane < 16) ? 0 : 8;      // A: 16x32 f16 K-interleave (ISA 7.12.2)
  const int bKb  = (lane < 16) ? 0 : 8;      // B: same K-interleave as A

  v8f acc = {};
  for (int k0 = 0; k0 < DCTX; k0 += 32) {
    v16h av, bv;
#pragma unroll
    for (int v = 0; v < 8; ++v) {
      int kk = (v < 4) ? (aKb + 2 * v) : (16 + aKb + 2 * (v - 4));
      av[2 * v]     = (_Float16)ctx[(m0 + Mrow) * DCTX + k0 + kk];
      av[2 * v + 1] = (_Float16)ctx[(m0 + Mrow) * DCTX + k0 + kk + 1];
    }
#pragma unroll
    for (int h = 0; h < 16; ++h)
      bv[h] = (_Float16)W[n * DCTX + k0 + bKb + h + ((h >= 8) ? 8 : 0)];     // B[k][n] = W[n][k]
    acc = __builtin_amdgcn_wmma_f32_16x16x32_f16(false, av, false, bv,
                                                 (short)0, acc, false, false);
  }
  const float bb = bias[n];
  const int   mb = (lane < 16) ? 0 : 8;                 // C/D: M = vgpr + 8*hi-half
#pragma unroll
  for (int v = 0; v < 8; ++v)
    angles[(m0 + mb + v) * NQ + n] = PI_F * tanhf(acc[v] + bb);
}

// ---------------------------------------------------------------------------
// Kernel 2: full 16-qubit statevector sim, entire state resident in LDS as
// half2 (256 KB), fp32 math per gate. One block (1024 thr / 32 wave32) per
// batch element.
//
// Optimizations:
//  * RZ*RY*RX fused into one 2x2 unitary per (layer,qubit); the batch-
//    dependent encoding RX is folded into layer 0 (angles add).
//  * The CNOT chain CNOT(0,1)..CNOT(14,15) is a GF(2)-linear permutation of
//    the index bits (suffix-XOR from MSB). It is applied SYMBOLICALLY:
//      Lrow[b]: logical bit b of amp at physical p = popc(p & Lrow[b]) & 1
//               (rows of L = phys->logical);  chain: Lrow[b] ^= Lrow[b+1] desc.
//      Pcol[b]: gate on logical bit b pairs physical {p, p ^ Pcol[b]}
//               (columns of Phi = L^-1);      chain: Pcol[b] ^= Pcol[b-1] desc.
//    => zero data-movement passes for CNOTs: 64 state passes total (was 124).
//  Qubit q <-> logical bit (15-q) (wire 0 = MSB).
// ---------------------------------------------------------------------------
struct cpx { float x, y; };
__device__ __forceinline__ cpx cmul(cpx a, cpx b) {
  return { a.x * b.x - a.y * b.y, a.x * b.y + a.y * b.x };
}
__device__ __forceinline__ cpx cadd(cpx a, cpx b) { return { a.x + b.x, a.y + b.y }; }

__global__ __launch_bounds__(NTHREADS)
void pqc_state_kernel(const float* __restrict__ angles,  // [512,16] from kernel 1
                      const float* __restrict__ prm,     // [4,16,3]
                      float* __restrict__ out)           // [512,16]
{
  extern __shared__ __half2 st[];          // 65536 * 4 B = 256 KB
  __shared__ float    sred[NQ];
  __shared__ unsigned Lrow[NQ];            // phys->logical bit masks
  __shared__ unsigned Pcol[NQ];            // logical->phys basis images

  const int tid = threadIdx.x;
  const int b   = blockIdx.x;

  // ---- init |0...0> and the identity bit-maps ----
  for (unsigned i = tid; i < NSTATE; i += NTHREADS)
    st[i] = __floats2half2_rn((i == 0) ? 1.0f : 0.0f, 0.0f);
  if (tid < NQ) {
    sred[tid] = 0.0f;
    Lrow[tid] = 1u << tid;
    Pcol[tid] = 1u << tid;
  }
  __syncthreads();

  // ---- circuit ----
  for (int l = 0; l < NLAYER; ++l) {
    for (int q = 0; q < NQ; ++q) {
      const int base = (l * NQ + q) * 3;
      float tx = prm[base + 0];
      if (l == 0) tx += angles[b * NQ + q];     // encoding RX folded in (angles add)
      const float ty = prm[base + 1];
      const float tz = prm[base + 2];

      // U = Rz(tz) * Ry(ty) * Rx(tx), built in fp32 (scalar, per thread)
      float sa, ca;  sincosf(0.5f * tx, &sa, &ca);
      float sb, cb;  sincosf(0.5f * ty, &sb, &cb);
      float sc, cc;  sincosf(0.5f * tz, &sc, &cc);
      cpx rx00{ca, 0.f}, rx01{0.f, -sa}, rx10{0.f, -sa}, rx11{ca, 0.f};
      cpx ry00{cb, 0.f}, ry01{-sb, 0.f}, ry10{sb, 0.f},  ry11{cb, 0.f};
      cpx m00 = cadd(cmul(ry00, rx00), cmul(ry01, rx10));
      cpx m01 = cadd(cmul(ry00, rx01), cmul(ry01, rx11));
      cpx m10 = cadd(cmul(ry10, rx00), cmul(ry11, rx10));
      cpx m11 = cadd(cmul(ry10, rx01), cmul(ry11, rx11));
      cpx rz0{cc, -sc}, rz1{cc, sc};
      const cpx u00 = cmul(rz0, m00), u01 = cmul(rz0, m01);
      const cpx u10 = cmul(rz1, m10), u11 = cmul(rz1, m11);

      const unsigned bq   = 15u - (unsigned)q;   // logical bit
      const unsigned mask = Pcol[bq];            // physical XOR-pair mask
      const unsigned sel  = Lrow[bq];            // parity mask: which side is |0>
      const unsigned s    = (unsigned)__builtin_ctz(mask);
      const unsigned lm   = (1u << s) - 1u;

      for (unsigned m = tid; m < NPAIR; m += NTHREADS) {
        const unsigned p0  = ((m & ~lm) << 1) | (m & lm);   // bit s clear
        const unsigned p1  = p0 ^ mask;                     // bit s set
        const unsigned par = (unsigned)__popc(p0 & sel) & 1u;
        const unsigned iA  = par ? p1 : p0;                 // logical |0> side
        const unsigned iB  = par ? p0 : p1;                 // logical |1> side
        const float2 a0 = __half22float2(st[iA]);
        const float2 a1 = __half22float2(st[iB]);
        const float n0x = u00.x * a0.x - u00.y * a0.y + u01.x * a1.x - u01.y * a1.y;
        const float n0y = u00.x * a0.y + u00.y * a0.x + u01.x * a1.y + u01.y * a1.x;
        const float n1x = u10.x * a0.x - u10.y * a0.y + u11.x * a1.x - u11.y * a1.y;
        const float n1y = u10.x * a0.y + u10.y * a0.x + u11.x * a1.y + u11.y * a1.x;
        st[iA] = __floats2half2_rn(n0x, n0y);
        st[iB] = __floats2half2_rn(n1x, n1y);
      }
      __syncthreads();
    }

    // ---- CNOT chain, applied symbolically to the bit-maps (thread 0) ----
    if (tid == 0) {
      // L_new = P o L : row_b = XOR_{j>=b} row_j  (suffix-XOR, descending)
      for (int bb = 14; bb >= 0; --bb) Lrow[bb] ^= Lrow[bb + 1];
      // Phi_new = Phi o P^-1 : col_b ^= col_{b-1} (descending; col_0 fixed)
      for (int bb = 15; bb >= 1; --bb) Pcol[bb] ^= Pcol[bb - 1];
    }
    __syncthreads();
  }

  // ---- <Z_q> : logical bit (15-q) of amp at p = popc(p & Lrow[15-q]) & 1 ----
  unsigned lr[NQ];
#pragma unroll
  for (int q = 0; q < NQ; ++q) lr[q] = Lrow[15 - q];

  float zacc[NQ];
#pragma unroll
  for (int q = 0; q < NQ; ++q) zacc[q] = 0.0f;

  for (unsigned i = tid; i < NSTATE; i += NTHREADS) {
    const float2 a = __half22float2(st[i]);
    const float p = a.x * a.x + a.y * a.y;
#pragma unroll
    for (int q = 0; q < NQ; ++q)
      zacc[q] += ((unsigned)__popc(i & lr[q]) & 1u) ? -p : p;
  }

  // wave32 reduction, then one LDS atomic per wave per qubit
#pragma unroll
  for (int q = 0; q < NQ; ++q) {
#pragma unroll
    for (int off = 16; off > 0; off >>= 1)
      zacc[q] += __shfl_xor(zacc[q], off, 32);
  }
  if ((tid & 31) == 0) {
#pragma unroll
    for (int q = 0; q < NQ; ++q)
      atomicAdd(&sred[q], zacc[q]);
  }
  __syncthreads();
  if (tid < NQ) out[b * NQ + tid] = sred[tid];
}

// ---------------------------------------------------------------------------
extern "C" void kernel_launch(void* const* d_in, const int* in_sizes, int n_in,
                              void* d_out, int out_size, void* d_ws, size_t ws_size,
                              hipStream_t stream)
{
  const float* ctx  = (const float*)d_in[0];   // [512,256]
  const float* W    = (const float*)d_in[1];   // [16,256]
  const float* bias = (const float*)d_in[2];   // [16]
  const float* prm  = (const float*)d_in[3];   // [4,16,3]
  float* out    = (float*)d_out;               // [512,16]
  float* angles = (float*)d_ws;                // scratch: 512*16 floats

  angles_wmma_kernel<<<B_TOT / 16, 32, 0, stream>>>(ctx, W, bias, angles);

  const size_t shmem = (size_t)NSTATE * sizeof(__half2);   // 256 KB (<= 320 KB/WGP)
  (void)hipFuncSetAttribute((const void*)pqc_state_kernel,
                            hipFuncAttributeMaxDynamicSharedMemorySize,
                            (int)shmem);
  pqc_state_kernel<<<B_TOT, NTHREADS, shmem, stream>>>(angles, prm, out);
}
